// LeakyRNNCell2_79774722555939
// MI455X (gfx1250) — matrix operand order repaired and checked by reading, not compiled, not run on hardware
//
#include <hip/hip_runtime.h>

// ---------------------------------------------------------------------------
// Leaky RNN scan (Yang-style) for MI455X / gfx1250.
//   pre = [x_t, h] @ W + b ;  h = 0.8*h + 0.2*relu(pre)
// Persistent-scan kernel: each 256-thread workgroup (8 wave32) owns 16 batch
// rows for all 205 timesteps. Recurrent GEMM uses v_wmma_f32_16x16x32_bf16
// with f32 accumulators; h state lives in f32 registers in C/D layout; the
// bf16 W^T stays register-resident per wave across the whole scan.
// Per step: ONE workgroup barrier (ping-pong z buffer), x_{t+1} prefetched
// into registers under the WMMA section.
// ---------------------------------------------------------------------------

typedef __attribute__((ext_vector_type(8)))  float  v8f;
typedef __attribute__((ext_vector_type(8)))  __bf16 v8bf;
typedef __attribute__((ext_vector_type(16))) __bf16 v16bf;

#define T_STEPS 205
#define BATCH   2048
#define N_IN    33
#define N_RNN   256
#define K_TOT   289          // N_IN + N_RNN
#define K_PAD   320          // padded K, multiple of 32 (pad region is zeros)
#define ALPHA   0.2f

#define M_WG     16          // batch rows per workgroup
#define NTHREADS 256         // 8 wave32
#define Z_STRIDE 328         // bf16 elems per z row; multiple of 8 -> 16B rows
#define X_ELEMS  (M_WG * N_IN)   // 528 staged x elements per step
#define XPJ      3           // ceil(528 / 256) prefetch regs per thread

// ---------------------------------------------------------------------------
// Pre-pass: W^T in bf16, padded. wt[n][k] = kernel[k][n] (k < 289), else 0.
// 256*320*2 = 160 KB -> workspace, resident in L2 (192 MB) for all 128 WGs.
// ---------------------------------------------------------------------------
__global__ void build_wt_kernel(const float* __restrict__ kernel,
                                __bf16* __restrict__ wt) {
  int idx = blockIdx.x * blockDim.x + threadIdx.x;
  if (idx >= N_RNN * K_PAD) return;
  int n = idx / K_PAD;
  int k = idx - n * K_PAD;
  float v = (k < K_TOT) ? kernel[k * N_RNN + n] : 0.0f;
  wt[idx] = (__bf16)v;
}

// ---------------------------------------------------------------------------
// Persistent scan kernel. Grid: 128 blocks x 256 threads (8 waves).
// Wave w computes output columns [32w, 32w+32) as 2 WMMA tiles of 16x16.
// ---------------------------------------------------------------------------
__global__ __launch_bounds__(NTHREADS) void leaky_rnn_scan_kernel(
    const float*  __restrict__ x,     // [T, B, N_IN]  f32
    const __bf16* __restrict__ wt,    // [N_RNN, K_PAD] bf16 (W^T, padded)
    const float*  __restrict__ bias,  // [N_RNN]
    const float*  __restrict__ h0,    // [B, N_RNN]
    float*        __restrict__ out)   // [T, B, N_RNN]
{
  // Ping-pong staging buffers: z[p] = [x_t | h_t | 0-pad] in bf16.
  __shared__ __attribute__((aligned(16))) __bf16 z[2][M_WG * Z_STRIDE];

  const int tid   = threadIdx.x;
  const int lane  = tid & 31;
  const int wave  = tid >> 5;        // 0..7
  const int row0  = blockIdx.x * M_WG;
  const int nlo   = lane & 15;       // A: row m; B: column n; C/D: column n
  const int khalf = lane >> 4;       // 0/1: K-half select per ISA layouts
  const int nbase = wave * 32;       // 2 tiles of 16 columns per wave

  // Zero both z buffers (pad region stays 0 forever).
  for (int i = tid; i < 2 * M_WG * Z_STRIDE; i += NTHREADS)
    z[0][i] = (__bf16)0.0f;          // flat over both buffers

  // Persistent h state in C/D register layout:
  //   lane l, vgpr g  <->  element (m = g + 8*(l>>4), n = nbase+16i+(l&15))
  v8f   hreg[2];
  float bvec[2];
#pragma unroll
  for (int i = 0; i < 2; ++i) {
    const int n = nbase + i * 16 + nlo;
    bvec[i] = bias[n];
#pragma unroll
    for (int g = 0; g < 8; ++g) {
      const int m = g + 8 * khalf;
      hreg[i][g] = h0[(size_t)(row0 + m) * N_RNN + n];
    }
  }

  // Seed z[0] h-region with h0 and x-region with x_0.
  for (int idx = tid; idx < M_WG * N_RNN; idx += NTHREADS) {
    const int m = idx >> 8;
    const int j = idx & 255;
    z[0][m * Z_STRIDE + N_IN + j] = (__bf16)h0[(size_t)(row0 + m) * N_RNN + j];
  }
  for (int idx = tid; idx < X_ELEMS; idx += NTHREADS) {
    const int m = idx / N_IN;
    const int k = idx - m * N_IN;
    z[0][m * Z_STRIDE + k] = (__bf16)x[(size_t)row0 * N_IN + idx - m * N_IN + (size_t)m * N_IN + k];
  }

  int p = 0;                          // buffer holding [x_t | h_t]

#pragma unroll 1
  for (int t = 0; t < T_STEPS; ++t) {
    // ---- Prefetch x_{t+1} into registers (latency hidden by the GEMM). ----
    const int tn = (t + 1 < T_STEPS) ? (t + 1) : t;   // clamp; last is unused
    const float* xnext = x + (size_t)tn * BATCH * N_IN + (size_t)row0 * N_IN;
    float xr[XPJ];
#pragma unroll
    for (int j = 0; j < XPJ; ++j) {
      const int idx = tid + j * NTHREADS;
      xr[j] = (idx < X_ELEMS) ? xnext[idx] : 0.0f;
    }

    __syncthreads();                 // publish z[p] (writes from iter t-1)

    // ---- GEMM: acc = [x_t | h_t] @ W  (bias pre-loaded into C). ----
    v8f acc[2];
#pragma unroll
    for (int i = 0; i < 2; ++i) {
#pragma unroll
      for (int g = 0; g < 8; ++g) acc[i][g] = bvec[i];
    }

    const __bf16* zb = &z[p][0];
#pragma unroll
    for (int kc = 0; kc < K_PAD; kc += 32) {
      // A fragment (16x32 bf16) — ISA 16-bit A layout:
      //   lanes 0-15 : elems 0-7 = K[kc..kc+7],  elems 8-15 = K[kc+16..+23]
      //   lanes 16-31: elems 0-7 = K[kc+8..+15], elems 8-15 = K[kc+24..+31]
      const __bf16* za = zb + nlo * Z_STRIDE + kc + khalf * 8;
      v8bf alo = *(const v8bf*)(za);
      v8bf ahi = *(const v8bf*)(za + 16);
      v16bf afrag = __builtin_shufflevector(alo, ahi,
          0, 1, 2, 3, 4, 5, 6, 7, 8, 9, 10, 11, 12, 13, 14, 15);

#pragma unroll
      for (int i = 0; i < 2; ++i) {
        // B fragment (32x16 bf16) — ISA 16-bit B layout (loop-invariant in t:
        // the compiler keeps all 20 fragments register-resident).
        const __bf16* wb = wt + (size_t)(nbase + i * 16 + nlo) * K_PAD
                              + kc + khalf * 8;
        v8bf blo = *(const v8bf*)(wb);
        v8bf bhi = *(const v8bf*)(wb + 16);
        v16bf bfrag = __builtin_shufflevector(blo, bhi,
            0, 1, 2, 3, 4, 5, 6, 7, 8, 9, 10, 11, 12, 13, 14, 15);

        acc[i] = __builtin_amdgcn_wmma_f32_16x16x32_bf16(
            /*neg_a=*/false, afrag, /*neg_b=*/false, bfrag,
            /*c_mod=*/(short)0, acc[i], /*reuse_a=*/false, /*reuse_b=*/false);
      }
    }

    // ---- Leaky-ReLU update; write h_{t+1} and x_{t+1} into z[p^1]. ----
    __bf16* zq = &z[p ^ 1][0];
    float* outt = out + (size_t)t * BATCH * N_RNN;
#pragma unroll
    for (int i = 0; i < 2; ++i) {
      const int n = nbase + i * 16 + nlo;
#pragma unroll
      for (int g = 0; g < 8; ++g) {
        const int m   = g + 8 * khalf;
        float pre = acc[i][g];                      // bias already included
        float act = pre > 0.0f ? pre : 0.0f;
        float hn  = (1.0f - ALPHA) * hreg[i][g] + ALPHA * act;
        hreg[i][g] = hn;
        outt[(size_t)(row0 + m) * N_RNN + n] = hn;
        zq[m * Z_STRIDE + N_IN + n] = (__bf16)hn;
      }
    }
#pragma unroll
    for (int j = 0; j < XPJ; ++j) {
      const int idx = tid + j * NTHREADS;
      if (idx < X_ELEMS) {
        const int m = idx / N_IN;
        const int k = idx - m * N_IN;
        zq[m * Z_STRIDE + k] = (__bf16)xr[j];
      }
    }
    p ^= 1;
    // Next iteration's single __syncthreads publishes z[p].
  }
}

// ---------------------------------------------------------------------------
// Harness entry point.
// Inputs (setup_inputs order): x [T,B,33] f32, kernel [289,256] f32,
//                              bias [256] f32, h0 [B,256] f32.
// Output: hs [T,B,256] f32.  Workspace: 160 KB for bf16 W^T.
// ---------------------------------------------------------------------------
extern "C" void kernel_launch(void* const* d_in, const int* in_sizes, int n_in,
                              void* d_out, int out_size, void* d_ws, size_t ws_size,
                              hipStream_t stream) {
  const float* x      = (const float*)d_in[0];
  const float* kernel = (const float*)d_in[1];
  const float* bias   = (const float*)d_in[2];
  const float* h0     = (const float*)d_in[3];
  float*       out    = (float*)d_out;
  __bf16*      wt     = (__bf16*)d_ws;   // N_RNN * K_PAD bf16 = 160 KB

  (void)in_sizes; (void)n_in; (void)out_size; (void)ws_size;

  // Pre-pass: transpose + pad + bf16-convert the weight matrix.
  {
    const int total = N_RNN * K_PAD;
    const int block = 256;
    const int grid  = (total + block - 1) / block;
    build_wt_kernel<<<grid, block, 0, stream>>>(kernel, wt);
  }

  // Persistent scan: 2048 rows / 16 per WG = 128 workgroups of 8 waves.
  leaky_rnn_scan_kernel<<<BATCH / M_WG, NTHREADS, 0, stream>>>(x, wt, bias, h0, out);
}
